// MessagePassingPathPlanningNetwork_42030549959248
// MI455X (gfx1250) — compile-verified, matrix operand order checked by reading, not run on hardware
//
#include <hip/hip_runtime.h>

// MessagePassing GNN forward for MI455X (gfx1250, wave32, WMMA).
// f32 activations/stats; bf16 WMMA GEMMs (f32 accumulate); weights pre-packed
// into per-lane fragment layout so B loads are single coalesced 32B loads.

typedef __attribute__((ext_vector_type(16))) __bf16 v16bf;
typedef __attribute__((ext_vector_type(8)))  __bf16 v8bf;
typedef __attribute__((ext_vector_type(8)))  float  v8f;

// A/B fragment K index for 16-bit 16x32 WMMA operands:
// vector elems 0..7  -> K = 8*khalf + 0..7
// vector elems 8..15 -> K = 16 + 8*khalf + 0..7
__device__ __forceinline__ int akmap(int i, int khalf) {
  return i + ((i >= 8) ? 8 : 0) + 8 * khalf;
}

// ------------------------------------------------ pack W (KxNC f32, row major)
// into bf16 fragments: out[((kb*NC/16 + nt)*32 + lane)*16 + i]
__global__ void pack_wfrag_kernel(const float* __restrict__ W, __bf16* __restrict__ out,
                                  int K, int NC) {
  int idx = blockIdx.x * blockDim.x + threadIdx.x;
  int total = (K / 32) * (NC / 16) * 512;
  if (idx >= total) return;
  int i = idx & 15, lane = (idx >> 4) & 31, frag = idx >> 9;
  int ntc = NC / 16;
  int kb = frag / ntc, nt = frag - kb * ntc;
  int khalf = lane >> 4;
  int k = kb * 32 + akmap(i, khalf);
  int col = nt * 16 + (lane & 15);
  out[idx] = (__bf16)W[k * NC + col];
}

// ---------------------------------------------------------------- input proj
__global__ void input_gemm_kernel(const float* __restrict__ x,
                                  const float* __restrict__ W,   // 18x64
                                  const float* __restrict__ b,
                                  float* __restrict__ h, int N) {
  int idx = blockIdx.x * blockDim.x + threadIdx.x;
  if (idx >= N * 64) return;
  int n = idx >> 6, c = idx & 63;
  float acc = b[c];
  const float* xr = x + (size_t)n * 18;
#pragma unroll
  for (int k = 0; k < 18; ++k) acc += xr[k] * W[k * 64 + c];
  h[idx] = acc;
}

__global__ void zero_kernel(float* __restrict__ p, int n) {
  int i = blockIdx.x * blockDim.x + threadIdx.x;
  if (i < n) p[i] = 0.f;
}

// sums: sum[64], sumsq[64] -> ss: scale[64], shift[64]
__global__ void bn_finalize_kernel(const float* __restrict__ sums,
                                   const float* __restrict__ g,
                                   const float* __restrict__ be,
                                   float cntInv, float* __restrict__ ss) {
  int c = threadIdx.x;
  if (c >= 64) return;
  float mean = sums[c] * cntInv;
  float var  = sums[64 + c] * cntInv - mean * mean;
  float sc   = g[c] * rsqrtf(var + 1e-5f);
  ss[c]      = sc;
  ss[64 + c] = be[c] - mean * sc;
}

// ------------------------- z = bf16(relu(scale*y + shift)), row-major (rows x 64)
__global__ void bn_relu_bf16_kernel(const float* __restrict__ y,
                                    const float* __restrict__ ss,
                                    __bf16* __restrict__ z, int quads) {
  int idx = blockIdx.x * blockDim.x + threadIdx.x;   // total/4 quads
  if (idx >= quads) return;
  int c = (idx * 4) & 63;
  float4 v = ((const float4*)y)[idx];
  __bf16 o0 = (__bf16)fmaxf(ss[c + 0] * v.x + ss[64 + c + 0], 0.f);
  __bf16 o1 = (__bf16)fmaxf(ss[c + 1] * v.y + ss[64 + c + 1], 0.f);
  __bf16 o2 = (__bf16)fmaxf(ss[c + 2] * v.z + ss[64 + c + 2], 0.f);
  __bf16 o3 = (__bf16)fmaxf(ss[c + 3] * v.w + ss[64 + c + 3], 0.f);
  __bf16* zp = z + (size_t)idx * 4;
  zp[0] = o0; zp[1] = o1; zp[2] = o2; zp[3] = o3;
}

// -------------------------------------------------- edge MLP linear 1 (K=129)
// A row e = [h[dst[e]](64) | h[src[e]](64)], plus rank-1 edge_attr term.
__global__ void edge_gemm1_kernel(const float* __restrict__ h,
                                  const float* __restrict__ ea,
                                  const int* __restrict__ ei, int E,
                                  const __bf16* __restrict__ wf,   // 4kb x 4nt frags
                                  const float* __restrict__ W1,    // 129x64 (row 128)
                                  const float* __restrict__ b1,
                                  float* __restrict__ yE,
                                  float* __restrict__ sums,
                                  int ntiles) {
  __shared__ float s_acc[128];
  const int tid = threadIdx.x;
  for (int i = tid; i < 128; i += blockDim.x) s_acc[i] = 0.f;
  __syncthreads();

  const int lane  = tid & 31;
  const int wave  = tid >> 5;
  const int gwave = blockIdx.x * (blockDim.x >> 5) + wave;
  const int nw    = gridDim.x * (blockDim.x >> 5);
  const int arow  = lane & 15;
  const int khalf = lane >> 4;
  const int ccol  = lane & 15;
  const int crb   = khalf * 8;
  const int* __restrict__ dstI = ei + E;

  for (int tile = gwave; tile < ntiles; tile += nw) {   // wave-uniform
    const int e0 = tile * 16;
    const int eA = e0 + arow;
    const float* hd = h + (size_t)dstI[eA] * 64;
    const float* hs = h + (size_t)ei[eA] * 64;
    v16bf A[4];
#pragma unroll
    for (int kb = 0; kb < 4; ++kb)
#pragma unroll
      for (int i = 0; i < 16; ++i) {
        int k = kb * 32 + akmap(i, khalf);
        float v = (k < 64) ? hd[k] : hs[k - 64];
        A[kb][i] = (__bf16)v;
      }
    float eav[8];
#pragma unroll
    for (int i = 0; i < 8; ++i) eav[i] = ea[e0 + crb + i];

#pragma clang loop unroll(disable)
    for (int nt = 0; nt < 4; ++nt) {
      const int col = nt * 16 + ccol;
      const float wEA = W1[128 * 64 + col];   // edge_attr row (rank-1)
      const float bc  = b1[col];
      v8f acc;
#pragma unroll
      for (int i = 0; i < 8; ++i) acc[i] = bc + eav[i] * wEA;
#pragma unroll
      for (int kb = 0; kb < 4; ++kb) {
        v16bf B = *(const v16bf*)(wf + ((size_t)(kb * 4 + nt) * 32 + lane) * 16);
        acc = __builtin_amdgcn_wmma_f32_16x16x32_bf16(false, A[kb], false, B,
                                                      (short)0, acc, false, false);
      }
      float ps = 0.f, pq = 0.f;
#pragma unroll
      for (int i = 0; i < 8; ++i) {
        float v = acc[i];
        yE[(size_t)(e0 + crb + i) * 64 + col] = v;
        ps += v; pq += v * v;
      }
      atomicAdd(&s_acc[col], ps);
      atomicAdd(&s_acc[64 + col], pq);
    }
  }
  __syncthreads();
  for (int i = tid; i < 128; i += blockDim.x) atomicAdd(&sums[i], s_acc[i]);
}

// -------------------- linear-2: y = z @ W2 + b2, z is bf16 rows x 64, + BN2 stats
__global__ void gemm2_kernel(const __bf16* __restrict__ z,     // bn_relu'd, bf16
                             const __bf16* __restrict__ wf,    // 2kb x 4nt frags
                             const float* __restrict__ b2,
                             float* __restrict__ y,            // rows x 64 out
                             float* __restrict__ sums,
                             int ntiles) {
  __shared__ float s_acc[128];
  const int tid = threadIdx.x;
  for (int i = tid; i < 128; i += blockDim.x) s_acc[i] = 0.f;
  __syncthreads();

  const int lane  = tid & 31;
  const int wave  = tid >> 5;
  const int gwave = blockIdx.x * (blockDim.x >> 5) + wave;
  const int nw    = gridDim.x * (blockDim.x >> 5);
  const int arow  = lane & 15;
  const int khalf = lane >> 4;
  const int ccol  = lane & 15;
  const int crb   = khalf * 8;

  for (int tile = gwave; tile < ntiles; tile += nw) {
    const int r0 = tile * 16;
    const __bf16* zr = z + (size_t)(r0 + arow) * 64;
    v16bf A[2];
#pragma unroll
    for (int kb = 0; kb < 2; ++kb) {
      int k0 = kb * 32 + 8 * khalf;
      v8bf lo = *(const v8bf*)(zr + k0);
      v8bf hi = *(const v8bf*)(zr + k0 + 16);
#pragma unroll
      for (int i = 0; i < 8; ++i) { A[kb][i] = lo[i]; A[kb][i + 8] = hi[i]; }
    }
#pragma clang loop unroll(disable)
    for (int nt = 0; nt < 4; ++nt) {
      const int col = nt * 16 + ccol;
      const float bc = b2[col];
      v8f acc;
#pragma unroll
      for (int i = 0; i < 8; ++i) acc[i] = bc;
#pragma unroll
      for (int kb = 0; kb < 2; ++kb) {
        v16bf B = *(const v16bf*)(wf + ((size_t)(kb * 4 + nt) * 32 + lane) * 16);
        acc = __builtin_amdgcn_wmma_f32_16x16x32_bf16(false, A[kb], false, B,
                                                      (short)0, acc, false, false);
      }
      float ps = 0.f, pq = 0.f;
#pragma unroll
      for (int i = 0; i < 8; ++i) {
        float v = acc[i];
        y[(size_t)(r0 + crb + i) * 64 + col] = v;
        ps += v; pq += v * v;
      }
      atomicAdd(&s_acc[col], ps);
      atomicAdd(&s_acc[64 + col], pq);
    }
  }
  __syncthreads();
  for (int i = tid; i < 128; i += blockDim.x) atomicAdd(&sums[i], s_acc[i]);
}

// ------------------------------------ scatter: agg[dst] += bn_relu(yE) rows
__global__ void edge_scatter_kernel(const float* __restrict__ yE,
                                    const float* __restrict__ ss,
                                    const int* __restrict__ dstI,
                                    float* __restrict__ agg, int E) {
  int idx = blockIdx.x * blockDim.x + threadIdx.x;   // E*16 quads
  if (idx >= E * 16) return;
  int e = idx >> 4, q = idx & 15;
  int d = dstI[e];
  float4 vv = ((const float4*)(yE + (size_t)e * 64))[q];
  float vals[4] = {vv.x, vv.y, vv.z, vv.w};
#pragma unroll
  for (int j = 0; j < 4; ++j) {
    int c = q * 4 + j;
    float v = fmaxf(ss[c] * vals[j] + ss[64 + c], 0.f);
    atomicAdd(&agg[(size_t)d * 64 + c], v);
  }
}

// ------------------------------ node MLP linear 1 (K=128): A = [h | agg]
__global__ void node_gemm1_kernel(const float* __restrict__ h,
                                  const float* __restrict__ agg,
                                  const __bf16* __restrict__ wf,  // 4kb x 4nt frags
                                  const float* __restrict__ b1,
                                  float* __restrict__ yN,
                                  float* __restrict__ sums,
                                  int ntiles) {
  __shared__ float s_acc[128];
  const int tid = threadIdx.x;
  for (int i = tid; i < 128; i += blockDim.x) s_acc[i] = 0.f;
  __syncthreads();

  const int lane  = tid & 31;
  const int wave  = tid >> 5;
  const int gwave = blockIdx.x * (blockDim.x >> 5) + wave;
  const int nw    = gridDim.x * (blockDim.x >> 5);
  const int arow  = lane & 15;
  const int khalf = lane >> 4;
  const int ccol  = lane & 15;
  const int crb   = khalf * 8;

  for (int tile = gwave; tile < ntiles; tile += nw) {
    const int n0 = tile * 16;
    const float* hr = h   + (size_t)(n0 + arow) * 64;
    const float* ar = agg + (size_t)(n0 + arow) * 64;
    v16bf A[4];
#pragma unroll
    for (int kb = 0; kb < 4; ++kb)
#pragma unroll
      for (int i = 0; i < 16; ++i) {
        int k = kb * 32 + akmap(i, khalf);
        float v = (k < 64) ? hr[k] : ar[k - 64];
        A[kb][i] = (__bf16)v;
      }
#pragma clang loop unroll(disable)
    for (int nt = 0; nt < 4; ++nt) {
      const int col = nt * 16 + ccol;
      const float bc = b1[col];
      v8f acc;
#pragma unroll
      for (int i = 0; i < 8; ++i) acc[i] = bc;
#pragma unroll
      for (int kb = 0; kb < 4; ++kb) {
        v16bf B = *(const v16bf*)(wf + ((size_t)(kb * 4 + nt) * 32 + lane) * 16);
        acc = __builtin_amdgcn_wmma_f32_16x16x32_bf16(false, A[kb], false, B,
                                                      (short)0, acc, false, false);
      }
      float ps = 0.f, pq = 0.f;
#pragma unroll
      for (int i = 0; i < 8; ++i) {
        float v = acc[i];
        yN[(size_t)(n0 + crb + i) * 64 + col] = v;
        ps += v; pq += v * v;
      }
      atomicAdd(&s_acc[col], ps);
      atomicAdd(&s_acc[64 + col], pq);
    }
  }
  __syncthreads();
  for (int i = tid; i < 128; i += blockDim.x) atomicAdd(&sums[i], s_acc[i]);
}

// ------------------------------- residual: h += bn_relu(yN)
__global__ void node_update_kernel(float* __restrict__ h,
                                   const float* __restrict__ yN,
                                   const float* __restrict__ ss, int total) {
  int idx = blockIdx.x * blockDim.x + threadIdx.x;
  if (idx >= total) return;
  int c = idx & 63;
  h[idx] += fmaxf(ss[c] * yN[idx] + ss[64 + c], 0.f);
}

// ---------------- fused prediction head: relu(h@predW + pb) @ poutW + pob
__global__ void pred_head_kernel(const float* __restrict__ h,
                                 const __bf16* __restrict__ predWf, // 2kb x 4nt
                                 const float* __restrict__ predB,
                                 const __bf16* __restrict__ poutWf, // 2kb x 1nt
                                 const float* __restrict__ poutB,
                                 float* __restrict__ out, int ntiles) {
  __shared__ float s_t[8][16 * 64];   // one 16x64 staging tile per wave
  const int tid   = threadIdx.x;
  const int lane  = tid & 31;
  const int wave  = tid >> 5;
  const int gwave = blockIdx.x * (blockDim.x >> 5) + wave;
  const int nw    = gridDim.x * (blockDim.x >> 5);
  const int arow  = lane & 15;
  const int khalf = lane >> 4;
  const int ccol  = lane & 15;
  const int crb   = khalf * 8;

  for (int tile = gwave; tile < ntiles; tile += nw) {
    const int n0 = tile * 16;
    const float* hr = h + (size_t)(n0 + arow) * 64;
    v16bf A[2];
#pragma unroll
    for (int kb = 0; kb < 2; ++kb)
#pragma unroll
      for (int i = 0; i < 16; ++i)
        A[kb][i] = (__bf16)hr[kb * 32 + akmap(i, khalf)];

#pragma clang loop unroll(disable)
    for (int nt = 0; nt < 4; ++nt) {
      const int col = nt * 16 + ccol;
      const float bc = predB[col];
      v8f acc;
#pragma unroll
      for (int i = 0; i < 8; ++i) acc[i] = bc;
#pragma unroll
      for (int kb = 0; kb < 2; ++kb) {
        v16bf B = *(const v16bf*)(predWf + ((size_t)(kb * 4 + nt) * 32 + lane) * 16);
        acc = __builtin_amdgcn_wmma_f32_16x16x32_bf16(false, A[kb], false, B,
                                                      (short)0, acc, false, false);
      }
#pragma unroll
      for (int i = 0; i < 8; ++i)
        s_t[wave][(crb + i) * 64 + col] = fmaxf(acc[i], 0.f);
    }
    asm volatile("" ::: "memory");   // per-wave LDS is in-order; pin compiler order

    v16bf A2[2];
#pragma unroll
    for (int kb = 0; kb < 2; ++kb)
#pragma unroll
      for (int i = 0; i < 16; ++i)
        A2[kb][i] = (__bf16)s_t[wave][arow * 64 + kb * 32 + akmap(i, khalf)];
    asm volatile("" ::: "memory");

    v8f acc;
    const float bo = poutB[ccol];
#pragma unroll
    for (int i = 0; i < 8; ++i) acc[i] = bo;
#pragma unroll
    for (int kb = 0; kb < 2; ++kb) {
      v16bf B = *(const v16bf*)(poutWf + ((size_t)kb * 32 + lane) * 16);
      acc = __builtin_amdgcn_wmma_f32_16x16x32_bf16(false, A2[kb], false, B,
                                                    (short)0, acc, false, false);
    }
#pragma unroll
    for (int i = 0; i < 8; ++i)
      out[(size_t)(n0 + crb + i) * 16 + ccol] = acc[i];
  }
}

static inline int imin_(int a, int b) { return a < b ? a : b; }

extern "C" void kernel_launch(void* const* d_in, const int* in_sizes, int n_in,
                              void* d_out, int out_size, void* d_ws, size_t ws_size,
                              hipStream_t stream) {
  const float* x         = (const float*)d_in[0];
  const float* edge_attr = (const float*)d_in[1];
  const float* W_in      = (const float*)d_in[2];
  const float* b_in      = (const float*)d_in[3];
  const float* msg_W1    = (const float*)d_in[4];
  const float* msg_b1    = (const float*)d_in[5];
  const float* msg_g1    = (const float*)d_in[6];
  const float* msg_be1   = (const float*)d_in[7];
  const float* msg_W2    = (const float*)d_in[8];
  const float* msg_b2    = (const float*)d_in[9];
  const float* msg_g2    = (const float*)d_in[10];
  const float* msg_be2   = (const float*)d_in[11];
  const float* upd_W1    = (const float*)d_in[12];
  const float* upd_b1    = (const float*)d_in[13];
  const float* upd_g1    = (const float*)d_in[14];
  const float* upd_be1   = (const float*)d_in[15];
  const float* upd_W2    = (const float*)d_in[16];
  const float* upd_b2    = (const float*)d_in[17];
  const float* upd_g2    = (const float*)d_in[18];
  const float* upd_be2   = (const float*)d_in[19];
  // 20..23: reward head — dead code in the reference return, skipped.
  const float* pred_W    = (const float*)d_in[24];
  const float* pred_b    = (const float*)d_in[25];
  const float* pout_W    = (const float*)d_in[26];
  const float* pout_b    = (const float*)d_in[27];
  const int*   edge_idx  = (const int*)d_in[28];
  float* out = (float*)d_out;

  const int N  = in_sizes[0] / 18;
  const int E  = in_sizes[1];
  const int NH = N * 64;
  const int NT_E = E / 16;   // 75000 (exact)
  const int NT_N = N / 16;   // 6250  (exact)

  // ws layout (f32 units): h | agg | yN | st(512) | wbf(51712) | yE(E*64) | zE(E*32)
  float*  h   = (float*)d_ws;
  float*  agg = h   + NH;
  float*  yN  = agg + NH;
  float*  st  = yN  + NH;           // sum1[128] ss1[128] sum2[128] ss2[128]
  __bf16* wbf = (__bf16*)(st + 512);
  float*  yE  = st + 512 + 51712;
  __bf16* zE  = (__bf16*)(yE + (size_t)E * 64);   // also reused for node z

  const int* dstI = edge_idx + E;

  // ---- pack all weights to bf16 fragment layout (tiny, L2-resident afterwards)
  for (int l = 0; l < 4; ++l) {
    __bf16* base = wbf + (size_t)l * 24576;
    pack_wfrag_kernel<<<32, 256, 0, stream>>>(msg_W1 + (size_t)l * 129 * 64, base, 128, 64);
    pack_wfrag_kernel<<<16, 256, 0, stream>>>(msg_W2 + (size_t)l * 64 * 64, base + 8192, 64, 64);
    pack_wfrag_kernel<<<32, 256, 0, stream>>>(upd_W1 + (size_t)l * 128 * 64, base + 12288, 128, 64);
    pack_wfrag_kernel<<<16, 256, 0, stream>>>(upd_W2 + (size_t)l * 64 * 64, base + 20480, 64, 64);
  }
  __bf16* predWf = wbf + 98304;
  __bf16* poutWf = wbf + 102400;
  pack_wfrag_kernel<<<16, 256, 0, stream>>>(pred_W, predWf, 64, 64);
  pack_wfrag_kernel<<<4, 256, 0, stream>>>(pout_W, poutWf, 64, 16);

  input_gemm_kernel<<<(NH + 255) / 256, 256, 0, stream>>>(x, W_in, b_in, h, N);

  const int gE = imin_((NT_E + 7) / 8, 2048);
  const int gN = imin_((NT_N + 7) / 8, 2048);
  const int EQ = E * 16;   // E*64/4 quads
  const int NQ = NH / 4;

  for (int l = 0; l < 4; ++l) {
    __bf16* eW1f = wbf + (size_t)l * 24576;
    __bf16* eW2f = eW1f + 8192;
    __bf16* nW1f = eW1f + 12288;
    __bf16* nW2f = eW1f + 20480;
    const float* mW1 = msg_W1 + (size_t)l * 129 * 64;
    const int lo = l * 64;

    // ---- edge MLP
    zero_kernel<<<2, 256, 0, stream>>>(st, 512);
    edge_gemm1_kernel<<<gE, 256, 0, stream>>>(h, edge_attr, edge_idx, E,
                                              eW1f, mW1, msg_b1 + lo, yE, st, NT_E);
    bn_finalize_kernel<<<1, 64, 0, stream>>>(st, msg_g1 + lo, msg_be1 + lo,
                                             1.0f / (float)E, st + 128);
    bn_relu_bf16_kernel<<<(EQ + 255) / 256, 256, 0, stream>>>(yE, st + 128, zE, EQ);
    gemm2_kernel<<<gE, 256, 0, stream>>>(zE, eW2f, msg_b2 + lo, yE, st + 256, NT_E);
    bn_finalize_kernel<<<1, 64, 0, stream>>>(st + 256, msg_g2 + lo, msg_be2 + lo,
                                             1.0f / (float)E, st + 384);
    zero_kernel<<<(NH + 255) / 256, 256, 0, stream>>>(agg, NH);
    edge_scatter_kernel<<<(EQ + 255) / 256, 256, 0, stream>>>(yE, st + 384,
                                                              dstI, agg, E);
    // ---- node MLP (residual)
    zero_kernel<<<2, 256, 0, stream>>>(st, 512);
    node_gemm1_kernel<<<gN, 256, 0, stream>>>(h, agg, nW1f, upd_b1 + lo,
                                              yN, st, NT_N);
    bn_finalize_kernel<<<1, 64, 0, stream>>>(st, upd_g1 + lo, upd_be1 + lo,
                                             1.0f / (float)N, st + 128);
    bn_relu_bf16_kernel<<<(NQ + 255) / 256, 256, 0, stream>>>(yN, st + 128, zE, NQ);
    gemm2_kernel<<<gN, 256, 0, stream>>>(zE, nW2f, upd_b2 + lo, yN, st + 256, NT_N);
    bn_finalize_kernel<<<1, 64, 0, stream>>>(st + 256, upd_g2 + lo, upd_be2 + lo,
                                             1.0f / (float)N, st + 384);
    node_update_kernel<<<(NH + 255) / 256, 256, 0, stream>>>(h, yN, st + 384, NH);
  }

  pred_head_kernel<<<gN, 256, 0, stream>>>(h, predWf, pred_b, poutWf, pout_b,
                                           out, NT_N);
}
